// SAGE_48524540510564
// MI455X (gfx1250) — matrix operand order repaired and checked by reading, not compiled, not run on hardware
//
#include <hip/hip_runtime.h>

// ---------------------------------------------------------------------------
// GraphSAGE 2-layer forward for MI455X (gfx1250, wave32, WMMA bf16)
//   layer: agg = scatter_mean(x[src] -> dst);  out = [agg|x] @ [Wl|Wr]^T + b
// ---------------------------------------------------------------------------

#define HD 128          // feature dim (IN == H == 128)

typedef __attribute__((ext_vector_type(16))) __bf16 v16bf;
typedef __attribute__((ext_vector_type(8)))  __bf16 v8bf;
typedef __attribute__((ext_vector_type(4)))  __bf16 v4bf;
typedef __attribute__((ext_vector_type(8)))  float  v8f;

// ------------------------------ utility kernels ----------------------------

__global__ void k_zero_f32(float* __restrict__ p, int n) {
  int i = blockIdx.x * blockDim.x + threadIdx.x;
  if (i < n) p[i] = 0.0f;
}

__global__ void k_cvt_f32_bf16(const float* __restrict__ in, __bf16* __restrict__ out, int n) {
  int i = blockIdx.x * blockDim.x + threadIdx.x;
  if (i < n) out[i] = (__bf16)in[i];
}

// ------------------------------ scatter (fp32 src) -------------------------
// one wave per edge: lane l moves features [4l, 4l+4) via float4 + 4 atomics.
__global__ void k_scatter_f32(const float* __restrict__ x,
                              const int* __restrict__ src, const int* __restrict__ dst,
                              float* __restrict__ agg, float* __restrict__ deg, int E) {
  int wave = (blockIdx.x * blockDim.x + threadIdx.x) >> 5;
  int lane = threadIdx.x & 31;
  if (wave >= E) return;
  int s = src[wave], d = dst[wave];
  float4 v = ((const float4*)(x + (size_t)s * HD))[lane];
  float* o = agg + (size_t)d * HD + lane * 4;
  atomicAdd(o + 0, v.x);
  atomicAdd(o + 1, v.y);
  atomicAdd(o + 2, v.z);
  atomicAdd(o + 3, v.w);
  if (lane == 0) atomicAdd(deg + d, 1.0f);
}

// ------------------------------ scatter (bf16 src) -------------------------
__global__ void k_scatter_bf16(const __bf16* __restrict__ h,
                               const int* __restrict__ src, const int* __restrict__ dst,
                               float* __restrict__ agg, int E) {
  int wave = (blockIdx.x * blockDim.x + threadIdx.x) >> 5;
  int lane = threadIdx.x & 31;
  if (wave >= E) return;
  int s = src[wave], d = dst[wave];
  v4bf v = ((const v4bf*)(h + (size_t)s * HD))[lane];
  float* o = agg + (size_t)d * HD + lane * 4;
  atomicAdd(o + 0, (float)v[0]);
  atomicAdd(o + 1, (float)v[1]);
  atomicAdd(o + 2, (float)v[2]);
  atomicAdd(o + 3, (float)v[3]);
}

// ------------------------------ normalize: agg/deg -> bf16 -----------------
// one wave per node; lane handles 4 features.
__global__ void k_norm(const float* __restrict__ agg, const float* __restrict__ deg,
                       __bf16* __restrict__ out, int N) {
  int wave = (blockIdx.x * blockDim.x + threadIdx.x) >> 5;
  int lane = threadIdx.x & 31;
  if (wave >= N) return;
  float inv = 1.0f / fmaxf(deg[wave], 1.0f);
  float4 v = ((const float4*)(agg + (size_t)wave * HD))[lane];
  v4bf o;
  o[0] = (__bf16)(v.x * inv);
  o[1] = (__bf16)(v.y * inv);
  o[2] = (__bf16)(v.z * inv);
  o[3] = (__bf16)(v.w * inv);
  ((v4bf*)(out + (size_t)wave * HD))[lane] = o;
}

// ------------------------------ WMMA GEMM ----------------------------------
// 4 k-chunks (K=128) of v_wmma_f32_16x16x32_bf16 accumulation.
__device__ inline v8f mma_quad(const __bf16* __restrict__ arow,   // row of A (len 128)
                               const __bf16* __restrict__ brow,   // W row slice (at 16*half)
                               int half, v8f c) {
#pragma unroll
  for (int kc = 0; kc < 4; ++kc) {
    int kb = kc * 32;
    union { v16bf v; v8bf h[2]; } a;
    a.h[0] = *(const v8bf*)(arow + kb + 16 * half);       // K run 1 (8 elems, 16B)
    a.h[1] = *(const v8bf*)(arow + kb + 16 * half + 8);   // K run 2 (8 elems, 16B)
    v16bf b = *(const v16bf*)(brow + kb);                 // 16 contiguous K (32B)
    c = __builtin_amdgcn_wmma_f32_16x16x32_bf16(false, a.v, false, b,
                                                (short)0, c, false, false);
  }
  return c;
}

// out = A1 @ W1^T + A2 @ W2^T + bias; mode 1: ReLU + dropout(0.5) -> bf16 outB
//                                    mode 0: fp32 -> outF
// One wave computes one 16x16 output tile; 8 col tiles per row tile, so the
// 8 waves of a block share one row tile's A data (L2/L0 friendly).
__global__ void k_gemm(const __bf16* __restrict__ A1, const __bf16* __restrict__ W1,
                       const __bf16* __restrict__ A2, const __bf16* __restrict__ W2,
                       const float* __restrict__ bias,
                       float* __restrict__ outF, __bf16* __restrict__ outB,
                       int mode, int Nrows) {
  int lane = threadIdx.x & 31;
  int waveLinear = (blockIdx.x * blockDim.x + threadIdx.x) >> 5;
  int rowTile = waveLinear >> 3;          // HD/16 == 8 column tiles
  int colTile = waveLinear & 7;
  int rowBase = rowTile * 16;
  if (rowBase >= Nrows) return;
  int half = lane >> 4;                   // 0: lanes 0-15, 1: lanes 16-31
  int mn   = lane & 15;
  int colBase = colTile * 16;

  int arowIdx = rowBase + mn;             // A-frag: lane holds row m = lane&15
  if (arowIdx >= Nrows) arowIdx = Nrows - 1;   // clamp (N%16==0 here anyway)
  const __bf16* ar1 = A1 + (size_t)arowIdx * HD;
  const __bf16* ar2 = A2 + (size_t)arowIdx * HD;
  // B-frag: lane holds column n = lane&15, contiguous K at 16*half
  const __bf16* br1 = W1 + (size_t)(colBase + mn) * HD + 16 * half;
  const __bf16* br2 = W2 + (size_t)(colBase + mn) * HD + 16 * half;

  v8f c = {};
  c = mma_quad(ar1, br1, half, c);        // agg  @ Wl^T
  c = mma_quad(ar2, br2, half, c);        // x/h  @ Wr^T

  float bv = bias[colBase + mn];          // n = lane&15 fixed per lane
  if (mode == 0) {
#pragma unroll
    for (int r = 0; r < 8; ++r) {
      int m = rowBase + r + 8 * half;     // C layout: VGPR r -> M = r + 8*half
      if (m < Nrows)
        outF[(size_t)m * HD + colBase + mn] = c[r] + bv;
    }
  } else {
#pragma unroll
    for (int r = 0; r < 8; ++r) {
      int m = rowBase + r + 8 * half;
      if (m < Nrows) {
        float v = fmaxf(c[r] + bv, 0.0f); // ReLU
        unsigned idx = (unsigned)(m * HD + colBase + mn);
        unsigned hh = idx * 2654435761u + 0x9E3779B9u;   // deterministic dropout hash
        hh ^= hh >> 15; hh *= 0x2C1B3C6Du; hh ^= hh >> 12;
        v = (hh & 1u) ? v * 2.0f : 0.0f;  // p=0.5, scale 1/(1-p)
        outB[(size_t)m * HD + colBase + mn] = (__bf16)v;
      }
    }
  }
}

// ------------------------------ launch -------------------------------------

static inline size_t align256(size_t x) { return (x + 255) & ~(size_t)255; }

extern "C" void kernel_launch(void* const* d_in, const int* in_sizes, int n_in,
                              void* d_out, int out_size, void* d_ws, size_t ws_size,
                              hipStream_t stream) {
  const float* x    = (const float*)d_in[0];
  const int*   ei   = (const int*)  d_in[1];
  const float* Wl0  = (const float*)d_in[2];
  const float* bl0  = (const float*)d_in[3];
  const float* Wr0  = (const float*)d_in[4];
  const float* Wl1  = (const float*)d_in[5];
  const float* bl1  = (const float*)d_in[6];
  const float* Wr1  = (const float*)d_in[7];
  float* out = (float*)d_out;

  const int N = in_sizes[0] / HD;       // 50000
  const int E = in_sizes[1] / 2;        // 800000
  const int* src = ei;                  // edge_index row 0
  const int* dst = ei + E;              // edge_index row 1

  // workspace carve-up
  char* ws = (char*)d_ws;  size_t cur = 0;
  float*  agg  = (float*) (ws + cur); cur = align256(cur + (size_t)N * HD * 4);
  float*  deg  = (float*) (ws + cur); cur = align256(cur + (size_t)N * 4);
  __bf16* aggb = (__bf16*)(ws + cur); cur = align256(cur + (size_t)N * HD * 2);
  __bf16* xb   = (__bf16*)(ws + cur); cur = align256(cur + (size_t)N * HD * 2);
  __bf16* hb   = (__bf16*)(ws + cur); cur = align256(cur + (size_t)N * HD * 2);
  __bf16* w0l  = (__bf16*)(ws + cur); cur = align256(cur + (size_t)HD * HD * 2);
  __bf16* w0r  = (__bf16*)(ws + cur); cur = align256(cur + (size_t)HD * HD * 2);
  __bf16* w1l  = (__bf16*)(ws + cur); cur = align256(cur + (size_t)HD * HD * 2);
  __bf16* w1r  = (__bf16*)(ws + cur); cur = align256(cur + (size_t)HD * HD * 2);
  (void)ws_size; (void)n_in; (void)out_size;

  const int T = 256;                               // 8 waves / block
  auto blks = [](long long n, int t) { return (int)((n + t - 1) / t); };

  // prep: zero accumulators, convert x + weights to bf16
  k_zero_f32   <<<blks((long long)N * HD, T), T, 0, stream>>>(agg, N * HD);
  k_zero_f32   <<<blks(N, T),              T, 0, stream>>>(deg, N);
  k_cvt_f32_bf16<<<blks((long long)N * HD, T), T, 0, stream>>>(x,   xb,  N * HD);
  k_cvt_f32_bf16<<<blks(HD * HD, T), T, 0, stream>>>(Wl0, w0l, HD * HD);
  k_cvt_f32_bf16<<<blks(HD * HD, T), T, 0, stream>>>(Wr0, w0r, HD * HD);
  k_cvt_f32_bf16<<<blks(HD * HD, T), T, 0, stream>>>(Wl1, w1l, HD * HD);
  k_cvt_f32_bf16<<<blks(HD * HD, T), T, 0, stream>>>(Wr1, w1r, HD * HD);

  // ---- layer 1 ----
  k_scatter_f32<<<blks((long long)E * 32, T), T, 0, stream>>>(x, src, dst, agg, deg, E);
  k_norm       <<<blks((long long)N * 32, T), T, 0, stream>>>(agg, deg, aggb, N);
  {
    long long waves = (long long)((N + 15) / 16) * 8;
    k_gemm<<<blks(waves * 32, T), T, 0, stream>>>(aggb, w0l, xb, w0r, bl0,
                                                  nullptr, hb, /*mode=*/1, N);
  }

  // ---- layer 2 (deg is identical; reuse) ----
  k_zero_f32   <<<blks((long long)N * HD, T), T, 0, stream>>>(agg, N * HD);
  k_scatter_bf16<<<blks((long long)E * 32, T), T, 0, stream>>>(hb, src, dst, agg, E);
  k_norm       <<<blks((long long)N * 32, T), T, 0, stream>>>(agg, deg, aggb, N);
  {
    long long waves = (long long)((N + 15) / 16) * 8;
    k_gemm<<<blks(waves * 32, T), T, 0, stream>>>(aggb, w1l, hb, w1r, bl1,
                                                  out, nullptr, /*mode=*/0, N);
  }
}
